// EfficientSparseAttentionModule_28140625723610
// MI455X (gfx1250) — matrix operand order repaired and checked by reading, not compiled, not run on hardware
//
#include <hip/hip_runtime.h>

#define DIM_C 256
#define NPIX  2304   // 48*48
#define NB    8
#define KSTR  264    // padded LDS row stride (bf16 elems) for K tile: 528B -> conflict-free frag reads
#define VSTR  40     // padded LDS row stride for V tile: 80B -> conflict-free frag reads

typedef __attribute__((ext_vector_type(16))) __bf16 v16bf;
typedef __attribute__((ext_vector_type(8)))  __bf16 v8bf;
typedef __attribute__((ext_vector_type(2)))  __bf16 v2bf;
typedef __attribute__((ext_vector_type(8)))  float  v8f;
typedef __attribute__((ext_vector_type(4)))  int    v4i;
typedef __attribute__((address_space(1)))    v4i    v4i_g;   // global
typedef __attribute__((address_space(3)))    v4i    v4i_l;   // LDS

#if __has_builtin(__builtin_amdgcn_exp2f)
#define EXP2F(x) __builtin_amdgcn_exp2f(x)
#else
#define EXP2F(x) exp2f(x)
#endif

#if __has_builtin(__builtin_amdgcn_global_load_async_to_lds_b128)
#define HAVE_ASYNC 1
#else
#define HAVE_ASYNC 0
#endif

// Native clang fptrunc float->bfloat (RNE); lets the backend pick
// v_cvt_pk_bf16_f32 / v_cvt_bf16 hardware when available.
__device__ __forceinline__ __bf16 f2bf(float f) { return (__bf16)f; }

__device__ __forceinline__ v2bf pk2(float a, float b) {
#if __has_builtin(__builtin_amdgcn_cvt_pk_bf16_f32)
    return __builtin_bit_cast(v2bf, __builtin_amdgcn_cvt_pk_bf16_f32(a, b));
#else
    v2bf r; r[0] = f2bf(a); r[1] = f2bf(b); return r;
#endif
}

union V16U { v16bf v; v2bf p[8]; };
union V8U  { v8bf  v; v2bf p[4]; };

__device__ __forceinline__ v16bf cat16(v8bf lo, v8bf hi) {
    return __builtin_shufflevector(lo, hi, 0,1,2,3,4,5,6,7,8,9,10,11,12,13,14,15);
}

// 16B global -> LDS copy: async DMA on gfx1250 when the builtin exists
__device__ __forceinline__ void cp16(__bf16* dst_lds, const __bf16* src_glb) {
#if HAVE_ASYNC
    __builtin_amdgcn_global_load_async_to_lds_b128(
        (v4i_g*)(v4i*)src_glb,    // generic reinterpret (drops const) then -> AS1
        (v4i_l*)(v4i*)dst_lds,    // generic reinterpret then -> AS3
        0, 0);
#else
    *reinterpret_cast<uint4*>(dst_lds) = *reinterpret_cast<const uint4*>(src_glb);
#endif
}

// ---------------------------------------------------------------------------
// Kernel 1: QKV projection (one wave per 16x16 tile, 8 bf16 WMMAs over c=256).
// Stores Q,K,V as [n][c] bf16; softmax scale folded into Q.
// ---------------------------------------------------------------------------
__global__ void qkv_proj_kernel(const float* __restrict__ x,
                                const float* __restrict__ wq,
                                const float* __restrict__ wk,
                                const float* __restrict__ wv,
                                const float* __restrict__ bq,
                                const float* __restrict__ bk,
                                const float* __restrict__ bv,
                                __bf16* __restrict__ Qb,
                                __bf16* __restrict__ Kb,
                                __bf16* __restrict__ Vb)
{
    const int lane = threadIdx.x & 31;
    const int wave = blockIdx.x * (blockDim.x >> 5) + (threadIdx.x >> 5);
    int nt = wave % (NPIX / 16);
    int r  = wave / (NPIX / 16);
    int ot = r % (DIM_C / 16);  r /= (DIM_C / 16);
    int b  = r % NB;            r /= NB;
    const int proj = r;  // 0=Q, 1=K, 2=V

    const float* W    = (proj == 0) ? wq : (proj == 1) ? wk : wv;
    const float* bias = (proj == 0) ? bq : (proj == 1) ? bk : bv;
    __bf16* out       = (proj == 0) ? Qb : (proj == 1) ? Kb : Vb;
    const float scale = (proj == 0) ? 0.0625f : 1.0f;   // 256^-0.5 folded into Q

    const int l15 = lane & 15, half = lane >> 4;
    const int n0 = nt * 16, o0 = ot * 16;
    const float* xb = x + (size_t)b * DIM_C * NPIX;

    v8f acc = {};
    #pragma unroll
    for (int j = 0; j < DIM_C / 32; ++j) {
        const int k0 = j * 32;
        // A-frag: weight row o0+l15; lane<16: c=k0+0..15 ; lane>=16: +16
        const float* wr = W + (size_t)(o0 + l15) * DIM_C + k0 + half * 16;
        V16U ua;
        #pragma unroll
        for (int i = 0; i < 4; ++i) {
            ua.p[i]     = pk2(wr[2 * i],      wr[2 * i + 1]);
            ua.p[4 + i] = pk2(wr[8 + 2 * i], wr[9 + 2 * i]);
        }
        // B-frag: x column n0+l15 over channels (stride NPIX, lane-coalesced)
        const float* xc = xb + (size_t)(k0 + half * 16) * NPIX + (n0 + l15);
        V16U ub;
        #pragma unroll
        for (int i = 0; i < 8; ++i)
            ub.p[i] = pk2(xc[(size_t)(2 * i) * NPIX], xc[(size_t)(2 * i + 1) * NPIX]);
        acc = __builtin_amdgcn_wmma_f32_16x16x32_bf16(false, ua.v, false, ub.v,
                                                      (short)0, acc, false, false);
    }
    const int ob = o0 + half * 8;
    V8U pk;
    #pragma unroll
    for (int i = 0; i < 4; ++i)
        pk.p[i] = pk2((acc[2 * i]     + bias[ob + 2 * i])     * scale,
                      (acc[2 * i + 1] + bias[ob + 2 * i + 1]) * scale);
    __bf16* dst = out + ((size_t)b * NPIX + (n0 + l15)) * DIM_C + ob;
    *reinterpret_cast<v8bf*>(dst) = pk.v;
}

// ---------------------------------------------------------------------------
// Kernel 2: LDS-tiled transpose V[n][c] -> VT[c][n] (bf16).
// ---------------------------------------------------------------------------
__global__ void vtrans_kernel(const __bf16* __restrict__ Vb,
                              __bf16* __restrict__ VT)
{
    __shared__ unsigned short tile[32][33];
    const int b  = blockIdx.z;
    const int n0 = blockIdx.x * 32;
    const int c0 = blockIdx.y * 32;
    const int tx = threadIdx.x, ty = threadIdx.y;
    const unsigned short* src =
        reinterpret_cast<const unsigned short*>(Vb) + (size_t)b * NPIX * DIM_C;
    unsigned short* dst =
        reinterpret_cast<unsigned short*>(VT) + (size_t)b * DIM_C * NPIX;
    #pragma unroll
    for (int r = 0; r < 4; ++r)
        tile[ty + r * 8][tx] = src[(size_t)(n0 + ty + r * 8) * DIM_C + c0 + tx];
    __syncthreads();
    #pragma unroll
    for (int r = 0; r < 4; ++r)
        dst[(size_t)(c0 + ty + r * 8) * NPIX + n0 + tx] = tile[tx][ty + r * 8];
}

// ---------------------------------------------------------------------------
// Cooperative stage of one 32-key slab: K[m0..m0+31][0..255] and
// VT[0..255][m0..m0+31] into padded LDS tiles. 256 threads, 16B per copy.
// ---------------------------------------------------------------------------
__device__ __forceinline__ void stage_kv(const __bf16* __restrict__ Kb,
                                         const __bf16* __restrict__ VT,
                                         __bf16* kl, __bf16* vl,
                                         int b, int m0, int tid)
{
    const __bf16* ksrc = Kb + ((size_t)b * NPIX + m0) * DIM_C;
    #pragma unroll
    for (int p = 0; p < 4; ++p) {
        int k   = p * 256 + tid;   // 0..1023
        int row = k >> 5;          // 0..31
        int c   = (k & 31) * 8;    // 0..248
        cp16(kl + row * KSTR + c, ksrc + (size_t)row * DIM_C + c);
    }
    const __bf16* vsrc = VT + (size_t)b * DIM_C * NPIX + m0;
    #pragma unroll
    for (int p = 0; p < 4; ++p) {
        int k  = p * 256 + tid;    // 0..1023
        int o  = k >> 2;           // 0..255
        int mm = (k & 3) * 8;      // 0..24
        cp16(vl + o * VSTR + mm, vsrc + (size_t)o * NPIX + mm);
    }
}

// ---------------------------------------------------------------------------
// Kernel 3: fused flash attention. Block = 8 waves, one batch, 8 n-tiles;
// K/V slabs double-buffered in LDS (async global->LDS DMA). Each wave:
// 16 queries x all 256 channels, S^T trick for in-lane softmax and in-lane
// P A-frag construction.
// ---------------------------------------------------------------------------
__global__ void __launch_bounds__(256, 1)
attn_kernel(const __bf16* __restrict__ Qb,
            const __bf16* __restrict__ Kb,
            const __bf16* __restrict__ VT,
            float* __restrict__ out)
{
    __shared__ __bf16 Klds[2][32 * KSTR];    // 2 x 16.5 KB
    __shared__ __bf16 Vlds[2][256 * VSTR];   // 2 x 20 KB

    const int tid  = threadIdx.x;
    const int lane = tid & 31;
    const int wv   = tid >> 5;               // 0..7
    const int b    = blockIdx.x / 18;        // 144 blocks: 18 per batch
    const int nt   = (blockIdx.x % 18) * 8 + wv;
    const int n0   = nt * 16;
    const int l15  = lane & 15, half = lane >> 4;
    const float L2E = 1.44269504088896340736f;

    // Resident Q B-frags: col n0+l15, k = 8h+0..7 and 16+8h+0..7 per lane-half.
    const __bf16* Qrow = Qb + ((size_t)b * NPIX + n0 + l15) * DIM_C;
    v16bf qf[8];
    #pragma unroll
    for (int j = 0; j < 8; ++j) {
        const __bf16* p = Qrow + j * 32 + half * 8;
        qf[j] = cat16(*reinterpret_cast<const v8bf*>(p),
                      *reinterpret_cast<const v8bf*>(p + 16));
    }

    v8f acc[16];
    #pragma unroll
    for (int t = 0; t < 16; ++t) { v8f z = {}; acc[t] = z; }
    float run_max = -3.0e38f;
    float run_sum = 0.0f;

    stage_kv(Kb, VT, &Klds[0][0], &Vlds[0][0], b, 0, tid);

    for (int it = 0; it < NPIX / 32; ++it) {
        const int cur = it & 1;
#if HAVE_ASYNC
        asm volatile("s_wait_asynccnt 0x0" ::: "memory");
#endif
        __syncthreads();   // staged slab visible; all waves done with other buffer
        if (it + 1 < NPIX / 32)
            stage_kv(Kb, VT, &Klds[cur ^ 1][0], &Vlds[cur ^ 1][0], b, (it + 1) * 32, tid);

        const __bf16* klw = &Klds[cur][0];
        const __bf16* vlw = &Vlds[cur][0];

        // --- S^T tiles for keys m0..m0+15 (s0) and m0+16..m0+31 (s1) ---
        v8f s0 = {}, s1 = {};
        #pragma unroll
        for (int j = 0; j < 8; ++j) {
            const __bf16* kr0 = klw + l15 * KSTR + j * 32 + half * 8;
            v16bf a0 = cat16(*reinterpret_cast<const v8bf*>(kr0),
                             *reinterpret_cast<const v8bf*>(kr0 + 16));
            s0 = __builtin_amdgcn_wmma_f32_16x16x32_bf16(false, a0, false, qf[j],
                                                         (short)0, s0, false, false);
            const __bf16* kr1 = kr0 + 16 * KSTR;
            v16bf a1 = cat16(*reinterpret_cast<const v8bf*>(kr1),
                             *reinterpret_cast<const v8bf*>(kr1 + 16));
            s1 = __builtin_amdgcn_wmma_f32_16x16x32_bf16(false, a1, false, qf[j],
                                                         (short)0, s1, false, false);
        }

        // --- online softmax over 32 keys (per query column) ---
        float tm = s0[0];
        #pragma unroll
        for (int i = 0; i < 8; ++i) { tm = fmaxf(tm, s0[i]); tm = fmaxf(tm, s1[i]); }
        tm = fmaxf(tm, __shfl_xor(tm, 16, 32));
        const bool bump = tm > run_max;
        const float nm  = fmaxf(run_max, tm);
        const float corr = EXP2F((run_max - nm) * L2E);
        run_max = nm;

        float p0[8], p1[8], ls = 0.0f;
        #pragma unroll
        for (int i = 0; i < 8; ++i) {
            p0[i] = EXP2F((s0[i] - nm) * L2E);
            p1[i] = EXP2F((s1[i] - nm) * L2E);
            ls += p0[i] + p1[i];
        }
        run_sum = run_sum * corr + ls;

        // P A-frag: built entirely within-lane from the two S^T C/D tiles.
        V16U ap;
        #pragma unroll
        for (int i = 0; i < 4; ++i) {
            ap.p[i]     = pk2(p0[2 * i], p0[2 * i + 1]);
            ap.p[4 + i] = pk2(p1[2 * i], p1[2 * i + 1]);
        }

        // Rescale only when some query's max moved (uniform branch, EXEC intact).
        if (__ballot(bump) != 0ull) {
            #pragma unroll
            for (int t = 0; t < 16; ++t) {
                #pragma unroll
                for (int i = 0; i < 8; ++i) acc[t][i] *= corr;
            }
        }
        #pragma unroll
        for (int t = 0; t < 16; ++t) {
            const __bf16* vr = vlw + (t * 16 + l15) * VSTR + half * 8;
            v16bf bv = cat16(*reinterpret_cast<const v8bf*>(vr),
                             *reinterpret_cast<const v8bf*>(vr + 16));
            acc[t] = __builtin_amdgcn_wmma_f32_16x16x32_bf16(false, ap.v, false, bv,
                                                             (short)0, acc[t], false, false);
        }
    }

    // --- finalize: 1/sum per query, gathered to match the C/D row layout ---
    float tot = run_sum + __shfl_xor(run_sum, 16, 32);
    float inv = 1.0f / tot;
    float invq[8];
    #pragma unroll
    for (int i = 0; i < 8; ++i) invq[i] = __shfl(inv, half * 8 + i, 32);

    float* ob = out + (size_t)b * DIM_C * NPIX;
    #pragma unroll
    for (int t = 0; t < 16; ++t) {
        float* dp = ob + (size_t)(t * 16 + l15) * NPIX + n0 + half * 8;
        float4 lo, hi;
        lo.x = acc[t][0] * invq[0]; lo.y = acc[t][1] * invq[1];
        lo.z = acc[t][2] * invq[2]; lo.w = acc[t][3] * invq[3];
        hi.x = acc[t][4] * invq[4]; hi.y = acc[t][5] * invq[5];
        hi.z = acc[t][6] * invq[6]; hi.w = acc[t][7] * invq[7];
        reinterpret_cast<float4*>(dp)[0] = lo;
        reinterpret_cast<float4*>(dp)[1] = hi;
    }
}

// ---------------------------------------------------------------------------
extern "C" void kernel_launch(void* const* d_in, const int* in_sizes, int n_in,
                              void* d_out, int out_size, void* d_ws, size_t ws_size,
                              hipStream_t stream) {
    const float* x  = (const float*)d_in[0];
    const float* wq = (const float*)d_in[1];
    const float* wk = (const float*)d_in[2];
    const float* wv = (const float*)d_in[3];
    const float* bq = (const float*)d_in[4];
    const float* bk = (const float*)d_in[5];
    const float* bv = (const float*)d_in[6];
    float* out = (float*)d_out;

    const size_t elems = (size_t)NB * NPIX * DIM_C;  // 4,718,592
    __bf16* Qb = (__bf16*)d_ws;
    __bf16* Kb = Qb + elems;
    __bf16* Vb = Kb + elems;
    __bf16* VT = Vb + elems;   // ~37.7 MB of d_ws total

    const int waves1 = 3 * NB * (DIM_C / 16) * (NPIX / 16);  // 55296 waves
    qkv_proj_kernel<<<waves1 / 8, 256, 0, stream>>>(x, wq, wk, wv, bq, bk, bv,
                                                    Qb, Kb, Vb);
    vtrans_kernel<<<dim3(NPIX / 32, DIM_C / 32, NB), dim3(32, 8), 0, stream>>>(Vb, VT);
    attn_kernel<<<NB * 18, 256, 0, stream>>>(Qb, Kb, VT, out);   // 144 blocks x 8 waves
}
